// MapEncoder_37331855737370
// MI455X (gfx1250) — compile-verified, matrix operand order checked by reading, not run on hardware
//
#include <hip/hip_runtime.h>
#include <math.h>

#define HIDDEN 128
#define NH 8

typedef __attribute__((ext_vector_type(16))) _Float16 v16h;
typedef __attribute__((ext_vector_type(8)))  float    v8f;

#define PI_F  3.14159265358979323846f
#define TPI_F 6.28318530717958647692f

// ---------------- WMMA tile helpers (CDNA5 wave32 layouts) ----------------

// A tile: 16 rows x 32 K, fp16, row-major in LDS with leading dim `lda`.
// lanes 0-15: M=lane, K = {k0..k0+7, k0+16..k0+23}
// lanes 16-31: M=lane-16, K = {k0+8..k0+15, k0+24..k0+31}
__device__ __forceinline__ v16h load_a_tile(const _Float16* s, int lda, int lane, int k0) {
    int half = lane >> 4, m = lane & 15;
    const _Float16* row = s + m * lda + k0 + half * 8;
    v16h a;
#pragma unroll
    for (int i = 0; i < 8; i++) { a[i] = row[i]; a[i + 8] = row[i + 16]; }
    return a;
}

// Weights are pre-packed into B-operand layout: for tile t = nt*kblocks + kb,
// lane l holds its full v16h at Wp[(t*32 + l)*16 .. +15]  (contiguous 32 B per lane).
__device__ __forceinline__ v8f wmma_rowblock(const _Float16* A, int lda,
                                             const _Float16* Wp, int ntile, int K, int lane) {
    v8f acc = {};
    int kblocks = K >> 5;
    const v16h* bp = (const v16h*)Wp + ((size_t)ntile * kblocks) * 32 + lane;
    for (int kb = 0; kb < kblocks; kb++) {
        v16h a = load_a_tile(A, lda, lane, kb * 32);
        v16h b = bp[(size_t)kb * 32];
        acc = __builtin_amdgcn_wmma_f32_16x16x32_f16(false, a, false, b, (short)0, acc, false, false);
    }
    return acc;
}

// Parallel row-wise LayerNorm: 16 rows x [128] fp32 in LDS -> fp16 in LDS.
// 256 threads, 16 per row; butterfly reduce inside each 16-lane half-wave.
__device__ __forceinline__ void ln_rows_to_f16(const float* sx, _Float16* dst,
                                               const float* g, const float* b, int doRelu) {
    int tid = threadIdx.x;
    int r = tid >> 4, c0 = (tid & 15) * 8;
    const float* row = sx + r * HIDDEN;
    float s = 0.f, s2 = 0.f;
#pragma unroll
    for (int j = 0; j < 8; j++) { float x = row[c0 + j]; s += x; s2 += x * x; }
#pragma unroll
    for (int off = 1; off < 16; off <<= 1) {
        s  += __shfl_xor(s, off);
        s2 += __shfl_xor(s2, off);
    }
    float mu  = s * (1.0f / HIDDEN);
    float var = s2 * (1.0f / HIDDEN) - mu * mu;
    float inv = rsqrtf(fmaxf(var, 0.f) + 1e-5f);
    _Float16* drow = dst + r * HIDDEN;
#pragma unroll
    for (int j = 0; j < 8; j++) {
        int cc = c0 + j;
        float y = (row[cc] - mu) * inv * g[cc] + b[cc];
        if (doRelu) y = fmaxf(y, 0.f);
        drow[cc] = (_Float16)y;
    }
    __syncthreads();
}

// ---------------- utility kernels ----------------

// Pack fp32 weight W[K][N] (row-major, K%32==0, N%16==0) into WMMA B-operand
// tile layout: dst[((nt*kblocks + kb)*32 + lane)*16 + i].
__global__ void pack_b_kernel(const float* __restrict__ W, _Float16* __restrict__ dst,
                              int K, int N) {
    int idx = blockIdx.x * 256 + threadIdx.x;
    if (idx >= K * N) return;
    int i = idx & 15;
    int l = (idx >> 4) & 31;
    int t = idx >> 9;
    int kblocks = K >> 5;
    int nt = t / kblocks, kb = t - nt * kblocks;
    int half = l >> 4;
    int n = nt * 16 + (l & 15);
    int k = kb * 32 + half * 8 + (i < 8 ? i : 16 + (i - 8));
    dst[idx] = (_Float16)W[(size_t)k * N + n];
}

__global__ void fill_f32_kernel(float* p, float v, int n) {
    int i = blockIdx.x * 256 + threadIdx.x;
    if (i < n) p[i] = v;
}

__global__ void fill_u32_kernel(unsigned* p, unsigned v, int n) {
    int i = blockIdx.x * 256 + threadIdx.x;
    if (i < n) p[i] = v;
}

// ---------------- fused node encoders ----------------

// centerline nodes: feats(4) -> MLP2 -> LN(ln1 of c2l attn) -> k,v projections (fp16 out)
__global__ __launch_bounds__(256)
void c_encode_kernel(const float* __restrict__ c_feats, const float* __restrict__ c_heading,
                     const float* __restrict__ c_length, int NC,
                     const float* __restrict__ w1, const float* __restrict__ b1,
                     const float* __restrict__ g, const float* __restrict__ be,
                     const _Float16* __restrict__ w2p, const float* __restrict__ b2,
                     const float* __restrict__ ln1g, const float* __restrict__ ln1b,
                     const _Float16* __restrict__ wkp, const float* __restrict__ bk,
                     const _Float16* __restrict__ wvp, const float* __restrict__ bv,
                     _Float16* __restrict__ knode, _Float16* __restrict__ vnode) {
    __shared__ float sx[16 * HIDDEN];
    __shared__ _Float16 a1[16 * HIDDEN];
    __shared__ float feat[16][4];
    int row0 = blockIdx.x * 16, tid = threadIdx.x;
    if (tid < 16) {
        int gi = row0 + tid;
        float f0 = 0, f1 = 0, f2 = 0, f3 = 0;
        if (gi < NC) { f0 = c_feats[2 * gi]; f1 = c_feats[2 * gi + 1]; f2 = c_heading[gi]; f3 = c_length[gi]; }
        feat[tid][0] = f0; feat[tid][1] = f1; feat[tid][2] = f2; feat[tid][3] = f3;
    }
    __syncthreads();
    {   // layer1 (K=4, VALU)
        int r = tid & 15, cg = tid >> 4;
#pragma unroll
        for (int j = 0; j < 8; j++) {
            int c = cg * 8 + j;
            float acc = b1[c];
#pragma unroll
            for (int k = 0; k < 4; k++) acc += feat[r][k] * w1[k * HIDDEN + c];
            sx[r * HIDDEN + c] = acc;
        }
    }
    __syncthreads();
    ln_rows_to_f16(sx, a1, g, be, 1);
    int wid = tid >> 5, lane = tid & 31, n = lane & 15, hi = lane >> 4;
    {   // layer2 -> sx (fp32)
        v8f acc = wmma_rowblock(a1, HIDDEN, w2p, wid, HIDDEN, lane);
#pragma unroll
        for (int r = 0; r < 8; r++) {
            int m = r + hi * 8;
            sx[m * HIDDEN + wid * 16 + n] = acc[r] + b2[wid * 16 + n];
        }
    }
    __syncthreads();
    ln_rows_to_f16(sx, a1, ln1g, ln1b, 0);
    {   // k projection
        v8f acc = wmma_rowblock(a1, HIDDEN, wkp, wid, HIDDEN, lane);
#pragma unroll
        for (int r = 0; r < 8; r++) {
            int m = r + hi * 8, gi = row0 + m, col = wid * 16 + n;
            if (gi < NC) knode[(size_t)gi * HIDDEN + col] = (_Float16)(acc[r] + bk[col]);
        }
    }
    {   // v projection
        v8f acc = wmma_rowblock(a1, HIDDEN, wvp, wid, HIDDEN, lane);
#pragma unroll
        for (int r = 0; r < 8; r++) {
            int m = r + hi * 8, gi = row0 + m, col = wid * 16 + n;
            if (gi < NC) vnode[(size_t)gi * HIDDEN + col] = (_Float16)(acc[r] + bv[col]);
        }
    }
}

// lane nodes: feats(7) -> MLP2 -> fp32 embeddings
__global__ __launch_bounds__(256)
void l_embed_kernel(const float* __restrict__ l_feats, const float* __restrict__ l_length,
                    const float* __restrict__ l_heading, const float* __restrict__ l_isi,
                    const float* __restrict__ l_turn, const float* __restrict__ l_tc, int NL,
                    const float* __restrict__ w1, const float* __restrict__ b1,
                    const float* __restrict__ g, const float* __restrict__ be,
                    const _Float16* __restrict__ w2p, const float* __restrict__ b2,
                    float* __restrict__ out) {
    __shared__ float sx[16 * HIDDEN];
    __shared__ _Float16 a1[16 * HIDDEN];
    __shared__ float feat[16][7];
    int row0 = blockIdx.x * 16, tid = threadIdx.x;
    if (tid < 16) {
        int gi = row0 + tid;
        float f[7] = {0, 0, 0, 0, 0, 0, 0};
        if (gi < NL) {
            f[0] = l_feats[2 * gi]; f[1] = l_feats[2 * gi + 1]; f[2] = l_length[gi];
            f[3] = l_heading[gi]; f[4] = l_isi[gi]; f[5] = l_turn[gi]; f[6] = l_tc[gi];
        }
        for (int i = 0; i < 7; i++) feat[tid][i] = f[i];
    }
    __syncthreads();
    {
        int r = tid & 15, cg = tid >> 4;
#pragma unroll
        for (int j = 0; j < 8; j++) {
            int c = cg * 8 + j;
            float acc = b1[c];
#pragma unroll
            for (int k = 0; k < 7; k++) acc += feat[r][k] * w1[k * HIDDEN + c];
            sx[r * HIDDEN + c] = acc;
        }
    }
    __syncthreads();
    ln_rows_to_f16(sx, a1, g, be, 1);
    int wid = tid >> 5, lane = tid & 31, n = lane & 15, hi = lane >> 4;
    v8f acc = wmma_rowblock(a1, HIDDEN, w2p, wid, HIDDEN, lane);
#pragma unroll
    for (int r = 0; r < 8; r++) {
        int m = r + hi * 8, gi = row0 + m, col = wid * 16 + n;
        if (gi < NL) out[(size_t)gi * HIDDEN + col] = acc[r] + b2[col];
    }
}

// LN(x) then up to 3 projections: q (fp32), k (fp16), v (fp16)
__global__ __launch_bounds__(256)
void ln_proj_kernel(const float* __restrict__ x, int N,
                    const float* __restrict__ g, const float* __restrict__ b,
                    const _Float16* __restrict__ wq, const float* __restrict__ bq, float* __restrict__ qout,
                    const _Float16* __restrict__ wk, const float* __restrict__ bk, _Float16* __restrict__ kout,
                    const _Float16* __restrict__ wv, const float* __restrict__ bv, _Float16* __restrict__ vout) {
    __shared__ float sx[16 * HIDDEN];
    __shared__ _Float16 a1[16 * HIDDEN];
    int row0 = blockIdx.x * 16, tid = threadIdx.x;
    for (int i = tid; i < 16 * HIDDEN; i += 256) {
        int m = i >> 7, gi = row0 + m;
        sx[i] = (gi < N) ? x[(size_t)gi * HIDDEN + (i & 127)] : 0.f;
    }
    __syncthreads();
    ln_rows_to_f16(sx, a1, g, b, 0);
    int wid = tid >> 5, lane = tid & 31, n = lane & 15, hi = lane >> 4;
    if (qout) {
        v8f acc = wmma_rowblock(a1, HIDDEN, wq, wid, HIDDEN, lane);
#pragma unroll
        for (int r = 0; r < 8; r++) {
            int m = r + hi * 8, gi = row0 + m, col = wid * 16 + n;
            if (gi < N) qout[(size_t)gi * HIDDEN + col] = acc[r] + bq[col];
        }
    }
    if (kout) {
        v8f acc = wmma_rowblock(a1, HIDDEN, wk, wid, HIDDEN, lane);
#pragma unroll
        for (int r = 0; r < 8; r++) {
            int m = r + hi * 8, gi = row0 + m, col = wid * 16 + n;
            if (gi < N) kout[(size_t)gi * HIDDEN + col] = (_Float16)(acc[r] + bk[col]);
        }
    }
    if (vout) {
        v8f acc = wmma_rowblock(a1, HIDDEN, wv, wid, HIDDEN, lane);
#pragma unroll
        for (int r = 0; r < 8; r++) {
            int m = r + hi * 8, gi = row0 + m, col = wid * 16 + n;
            if (gi < N) vout[(size_t)gi * HIDDEN + col] = (_Float16)(acc[r] + bv[col]);
        }
    }
}

// ---------------- fused edge attr MLP2 + ke/ve projections + node gather-add ----------------

template <int DIN, bool L2L>
__global__ __launch_bounds__(256)
void edge_kv_kernel(const int* __restrict__ ei, int E,
                    const float* __restrict__ spos, const float* __restrict__ shead,
                    const float* __restrict__ dpos, const float* __restrict__ dhead,
                    const float* __restrict__ hop, const float* __restrict__ type3,
                    const float* __restrict__ w1, const float* __restrict__ b1,
                    const float* __restrict__ g, const float* __restrict__ be,
                    const _Float16* __restrict__ w2p, const float* __restrict__ b2,
                    const _Float16* __restrict__ wkep, const float* __restrict__ bke,
                    const _Float16* __restrict__ wvep, const float* __restrict__ bve,
                    const _Float16* __restrict__ knode, const _Float16* __restrict__ vnode,
                    _Float16* __restrict__ kfull, _Float16* __restrict__ vfull) {
    __shared__ float sx[16 * HIDDEN];
    __shared__ _Float16 a1[16 * HIDDEN];
    __shared__ _Float16 a2[16 * HIDDEN];
    __shared__ float feat[16][8];
    __shared__ int ssrc[16];
    int row0 = blockIdx.x * 16, tid = threadIdx.x;
    if (tid < 16) {
        int e = row0 + tid;
        float f[7] = {0, 0, 0, 0, 0, 0, 0};
        int s = 0;
        if (e < E) {
            s = ei[e];
            int d = ei[E + e];
            float vx = spos[2 * s] - dpos[2 * d];
            float vy = spos[2 * s + 1] - dpos[2 * d + 1];
            float hd = dhead[d];
            float ch = cosf(hd), sh = sinf(hd);
            float lx = ch * vx + sh * vy;
            float ly = -sh * vx + ch * vy;
            f[0] = sqrtf(lx * lx + ly * ly);
            f[1] = atan2f(ly, lx);
            float da = shead[s] - hd + PI_F;
            da = fmodf(da, TPI_F);
            if (da < 0.f) da += TPI_F;
            f[2] = da - PI_F;
            if (L2L) { f[3] = hop[e]; f[4] = type3[3 * e]; f[5] = type3[3 * e + 1]; f[6] = type3[3 * e + 2]; }
        }
        ssrc[tid] = s;
#pragma unroll
        for (int i = 0; i < DIN; i++) feat[tid][i] = f[i];
    }
    __syncthreads();
    {   // layer1
        int r = tid & 15, cg = tid >> 4;
#pragma unroll
        for (int j = 0; j < 8; j++) {
            int c = cg * 8 + j;
            float acc = b1[c];
#pragma unroll
            for (int k = 0; k < DIN; k++) acc += feat[r][k] * w1[k * HIDDEN + c];
            sx[r * HIDDEN + c] = acc;
        }
    }
    __syncthreads();
    ln_rows_to_f16(sx, a1, g, be, 1);
    int wid = tid >> 5, lane = tid & 31, n = lane & 15, hi = lane >> 4;
    {   // edge_attr = layer2 output (fp16 in a2)
        v8f acc = wmma_rowblock(a1, HIDDEN, w2p, wid, HIDDEN, lane);
#pragma unroll
        for (int r = 0; r < 8; r++) {
            int m = r + hi * 8;
            a2[m * HIDDEN + wid * 16 + n] = (_Float16)(acc[r] + b2[wid * 16 + n]);
        }
    }
    __syncthreads();
    {   // k_full = knode[src] + attr @ wke + bke
        v8f acc = wmma_rowblock(a2, HIDDEN, wkep, wid, HIDDEN, lane);
#pragma unroll
        for (int r = 0; r < 8; r++) {
            int m = r + hi * 8, e = row0 + m, col = wid * 16 + n;
            if (e < E) {
                float val = acc[r] + bke[col] + (float)knode[(size_t)ssrc[m] * HIDDEN + col];
                kfull[(size_t)e * HIDDEN + col] = (_Float16)val;
            }
        }
    }
    {   // v_full = vnode[src] + attr @ wve + bve
        v8f acc = wmma_rowblock(a2, HIDDEN, wvep, wid, HIDDEN, lane);
#pragma unroll
        for (int r = 0; r < 8; r++) {
            int m = r + hi * 8, e = row0 + m, col = wid * 16 + n;
            if (e < E) {
                float val = acc[r] + bve[col] + (float)vnode[(size_t)ssrc[m] * HIDDEN + col];
                vfull[(size_t)e * HIDDEN + col] = (_Float16)val;
            }
        }
    }
}

// ---------------- segment softmax passes ----------------

__global__ void score_kernel(const float* __restrict__ q, const _Float16* __restrict__ kfull,
                             const int* __restrict__ ei, int E,
                             float* __restrict__ sc, unsigned* __restrict__ mbuf) {
    int t = blockIdx.x * 256 + threadIdx.x;
    if (t >= E * NH) return;
    int e = t >> 3, h = t & 7;
    int d = ei[E + e];
    const float* qr = q + (size_t)d * HIDDEN + h * 16;
    const _Float16* kr = kfull + (size_t)e * HIDDEN + h * 16;
    float s = 0.f;
#pragma unroll
    for (int i = 0; i < 16; i++) s += qr[i] * (float)kr[i];
    s *= 0.25f;  // 1/sqrt(16)
    sc[(size_t)e * NH + h] = s;
    unsigned u = __float_as_uint(s);
    u = (u & 0x80000000u) ? ~u : (u | 0x80000000u);  // monotone map: float max via uint atomic
    atomicMax(&mbuf[d * NH + h], u);
}

__global__ void expsum_kernel(float* __restrict__ sc, const unsigned* __restrict__ mbuf,
                              const int* __restrict__ ei, int E, float* __restrict__ den) {
    int t = blockIdx.x * 256 + threadIdx.x;
    if (t >= E * NH) return;
    int e = t >> 3, h = t & 7;
    int d = ei[E + e];
    unsigned u = mbuf[d * NH + h];
    float m = (u & 0x80000000u) ? __uint_as_float(u ^ 0x80000000u) : __uint_as_float(~u);
    float w = expf(sc[(size_t)e * NH + h] - m);
    sc[(size_t)e * NH + h] = w;
    atomicAdd(&den[d * NH + h], w);
}

__global__ void agg_kernel(const float* __restrict__ sc, const float* __restrict__ den,
                           const _Float16* __restrict__ vfull, const int* __restrict__ ei, int E,
                           float* __restrict__ agg) {
    int t = blockIdx.x * 256 + threadIdx.x;
    if (t >= E * NH) return;
    int e = t >> 3, h = t & 7;
    int d = ei[E + e];
    float alpha = sc[(size_t)e * NH + h] / (den[d * NH + h] + 1e-16f);
    const _Float16* vr = vfull + (size_t)e * HIDDEN + h * 16;
    float* ar = agg + (size_t)d * HIDDEN + h * 16;
#pragma unroll
    for (int i = 0; i < 16; i++) atomicAdd(&ar[i], alpha * (float)vr[i]);
}

// ---------------- attention output projection + FFN (fused) ----------------

__global__ __launch_bounds__(256)
void attn_out_ffn_kernel(const float* __restrict__ xdst, const float* __restrict__ agg, int N,
                         const _Float16* __restrict__ wop, const float* __restrict__ bo,
                         const float* __restrict__ ln2g, const float* __restrict__ ln2b,
                         const _Float16* __restrict__ fw1p, const float* __restrict__ fb1,
                         const _Float16* __restrict__ fw2p, const float* __restrict__ fb2,
                         float* __restrict__ out) {
    __shared__ float sy[16 * HIDDEN];
    __shared__ _Float16 a1[16 * HIDDEN];
    __shared__ _Float16 hh[16 * 512];
    int row0 = blockIdx.x * 16, tid = threadIdx.x;
    for (int i = tid; i < 16 * HIDDEN; i += 256) {
        int m = i >> 7, gi = row0 + m;
        float v = (gi < N) ? agg[(size_t)gi * HIDDEN + (i & 127)] : 0.f;
        a1[i] = (_Float16)v;
    }
    __syncthreads();
    int wid = tid >> 5, lane = tid & 31, n = lane & 15, hi = lane >> 4;
    {   // out = x_dst + agg @ wo + bo
        v8f acc = wmma_rowblock(a1, HIDDEN, wop, wid, HIDDEN, lane);
#pragma unroll
        for (int r = 0; r < 8; r++) {
            int m = r + hi * 8, gi = row0 + m, col = wid * 16 + n;
            float xv = (gi < N) ? xdst[(size_t)gi * HIDDEN + col] : 0.f;
            sy[m * HIDDEN + col] = xv + acc[r] + bo[col];
        }
    }
    __syncthreads();
    ln_rows_to_f16(sy, a1, ln2g, ln2b, 0);
    // hidden = relu(LN(out) @ fw1 + fb1)  -> [16,512] fp16 in LDS
    for (int t4 = 0; t4 < 4; t4++) {
        int nt = wid * 4 + t4;
        int n0 = nt * 16;
        v8f acc = wmma_rowblock(a1, HIDDEN, fw1p, nt, HIDDEN, lane);
#pragma unroll
        for (int r = 0; r < 8; r++) {
            int m = r + hi * 8;
            float y = acc[r] + fb1[n0 + n];
            hh[m * 512 + n0 + n] = (_Float16)fmaxf(y, 0.f);
        }
    }
    __syncthreads();
    {   // final = out + hidden @ fw2 + fb2
        v8f acc = wmma_rowblock(hh, 512, fw2p, wid, 512, lane);
#pragma unroll
        for (int r = 0; r < 8; r++) {
            int m = r + hi * 8, gi = row0 + m, col = wid * 16 + n;
            if (gi < N) out[(size_t)gi * HIDDEN + col] = sy[m * HIDDEN + col] + acc[r] + fb2[col];
        }
    }
}

// ---------------- host orchestration ----------------

extern "C" void kernel_launch(void* const* d_in, const int* in_sizes, int n_in,
                              void* d_out, int out_size, void* d_ws, size_t ws_size,
                              hipStream_t stream) {
    (void)out_size; (void)ws_size;
    if (n_in < 79) return;

    auto F = [&](int i) { return (const float*)d_in[i]; };

    const float* c_position = F(0);
    const float* c_feats    = F(1);
    const float* c_heading  = F(2);
    const float* c_length   = F(3);
    const float* l_position = F(4);
    const float* l_feats    = F(5);
    const float* l_length   = F(6);
    const float* l_heading  = F(7);
    const float* l_isi      = F(8);
    const float* l_turn     = F(9);
    const float* l_tc       = F(10);
    const float* hop        = F(11);
    const float* type3      = F(12);
    const int* c2l_ei = (const int*)d_in[77];
    const int* l2l_ei = (const int*)d_in[78];

    const int NC = in_sizes[2];
    const int NL = in_sizes[7];
    const int E1 = in_sizes[77] / 2;
    const int E2 = in_sizes[78] / 2;
    const int Emax = (E1 > E2) ? E1 : E2;

    // param index bases (flattened dict order)
    const int CEMB = 13, LEMB = 19, C2LE = 25, L2LE = 31, A1 = 37, A2 = 57;
    // attn offsets: wq0 bq1 wk2 bk3 wv4 bv5 wke6 bke7 wve8 bve9 wo10 bo11
    //               ln1g12 ln1b13 ln2g14 ln2b15 fw1 16 fb1 17 fw2 18 fb2 19

    // workspace bump allocator
    char* base = (char*)d_ws;
    size_t off = 0;
    auto alloc = [&](size_t bytes) -> void* {
        void* p = base + off;
        off += (bytes + 255) & ~(size_t)255;
        return p;
    };
    auto pack = [&](int idx, int K, int N) -> _Float16* {
        _Float16* dst = (_Float16*)alloc((size_t)K * N * sizeof(_Float16));
        pack_b_kernel<<<(K * N + 255) / 256, 256, 0, stream>>>((const float*)d_in[idx], dst, K, N);
        return dst;
    };

    // ---- pack weights to fp16 B-operand layout ----
    _Float16* w2_cemb = pack(CEMB + 4, HIDDEN, HIDDEN);
    _Float16* w2_lemb = pack(LEMB + 4, HIDDEN, HIDDEN);
    _Float16* w2_c2le = pack(C2LE + 4, HIDDEN, HIDDEN);
    _Float16* w2_l2le = pack(L2LE + 4, HIDDEN, HIDDEN);
    _Float16* wq1p  = pack(A1 + 0, HIDDEN, HIDDEN);
    _Float16* wk1p  = pack(A1 + 2, HIDDEN, HIDDEN);
    _Float16* wv1p  = pack(A1 + 4, HIDDEN, HIDDEN);
    _Float16* wke1p = pack(A1 + 6, HIDDEN, HIDDEN);
    _Float16* wve1p = pack(A1 + 8, HIDDEN, HIDDEN);
    _Float16* wo1p  = pack(A1 + 10, HIDDEN, HIDDEN);
    _Float16* fw11p = pack(A1 + 16, HIDDEN, 512);
    _Float16* fw21p = pack(A1 + 18, 512, HIDDEN);
    _Float16* wq2p  = pack(A2 + 0, HIDDEN, HIDDEN);
    _Float16* wk2p  = pack(A2 + 2, HIDDEN, HIDDEN);
    _Float16* wv2p  = pack(A2 + 4, HIDDEN, HIDDEN);
    _Float16* wke2p = pack(A2 + 6, HIDDEN, HIDDEN);
    _Float16* wve2p = pack(A2 + 8, HIDDEN, HIDDEN);
    _Float16* wo2p  = pack(A2 + 10, HIDDEN, HIDDEN);
    _Float16* fw12p = pack(A2 + 16, HIDDEN, 512);
    _Float16* fw22p = pack(A2 + 18, 512, HIDDEN);

    // ---- activation buffers ----
    _Float16* knode_c = (_Float16*)alloc((size_t)NC * HIDDEN * 2);
    _Float16* vnode_c = (_Float16*)alloc((size_t)NC * HIDDEN * 2);
    _Float16* knode_l = (_Float16*)alloc((size_t)NL * HIDDEN * 2);
    _Float16* vnode_l = (_Float16*)alloc((size_t)NL * HIDDEN * 2);
    float* l_embs0 = (float*)alloc((size_t)NL * HIDDEN * 4);
    float* l_embs1 = (float*)alloc((size_t)NL * HIDDEN * 4);
    float* q_l     = (float*)alloc((size_t)NL * HIDDEN * 4);
    _Float16* kfull = (_Float16*)alloc((size_t)Emax * HIDDEN * 2);
    _Float16* vfull = (_Float16*)alloc((size_t)Emax * HIDDEN * 2);
    float* sc      = (float*)alloc((size_t)Emax * NH * 4);
    unsigned* mbuf = (unsigned*)alloc((size_t)NL * NH * 4);
    float* den     = (float*)alloc((size_t)NL * NH * 4);
    float* aggb    = (float*)alloc((size_t)NL * HIDDEN * 4);

    const int tilesC = (NC + 15) / 16;
    const int tilesL = (NL + 15) / 16;
    const int tilesE1 = (E1 + 15) / 16;
    const int tilesE2 = (E2 + 15) / 16;

    auto zero_accums = [&]() {
        fill_u32_kernel<<<(NL * NH + 255) / 256, 256, 0, stream>>>(mbuf, 0u, NL * NH);
        fill_f32_kernel<<<(NL * NH + 255) / 256, 256, 0, stream>>>(den, 0.f, NL * NH);
        fill_f32_kernel<<<(NL * HIDDEN + 255) / 256, 256, 0, stream>>>(aggb, 0.f, NL * HIDDEN);
    };

    // ======== stage 1: node encoders ========
    c_encode_kernel<<<tilesC, 256, 0, stream>>>(
        c_feats, c_heading, c_length, NC,
        F(CEMB + 0), F(CEMB + 1), F(CEMB + 2), F(CEMB + 3), w2_cemb, F(CEMB + 5),
        F(A1 + 12), F(A1 + 13),
        wk1p, F(A1 + 3), wv1p, F(A1 + 5),
        knode_c, vnode_c);

    l_embed_kernel<<<tilesL, 256, 0, stream>>>(
        l_feats, l_length, l_heading, l_isi, l_turn, l_tc, NL,
        F(LEMB + 0), F(LEMB + 1), F(LEMB + 2), F(LEMB + 3), w2_lemb, F(LEMB + 5),
        l_embs0);

    // ======== layer 1: c2l attention ========
    ln_proj_kernel<<<tilesL, 256, 0, stream>>>(
        l_embs0, NL, F(A1 + 12), F(A1 + 13),
        wq1p, F(A1 + 1), q_l,
        (const _Float16*)nullptr, (const float*)nullptr, (_Float16*)nullptr,
        (const _Float16*)nullptr, (const float*)nullptr, (_Float16*)nullptr);

    edge_kv_kernel<3, false><<<tilesE1, 256, 0, stream>>>(
        c2l_ei, E1, c_position, c_heading, l_position, l_heading,
        (const float*)nullptr, (const float*)nullptr,
        F(C2LE + 0), F(C2LE + 1), F(C2LE + 2), F(C2LE + 3), w2_c2le, F(C2LE + 5),
        wke1p, F(A1 + 7), wve1p, F(A1 + 9),
        knode_c, vnode_c, kfull, vfull);

    zero_accums();
    {
        int nt = E1 * NH, gb = (nt + 255) / 256;
        score_kernel<<<gb, 256, 0, stream>>>(q_l, kfull, c2l_ei, E1, sc, mbuf);
        expsum_kernel<<<gb, 256, 0, stream>>>(sc, mbuf, c2l_ei, E1, den);
        agg_kernel<<<gb, 256, 0, stream>>>(sc, den, vfull, c2l_ei, E1, aggb);
    }
    attn_out_ffn_kernel<<<tilesL, 256, 0, stream>>>(
        l_embs0, aggb, NL,
        wo1p, F(A1 + 11), F(A1 + 14), F(A1 + 15),
        fw11p, F(A1 + 17), fw21p, F(A1 + 19),
        l_embs1);

    // ======== layer 2: l2l attention ========
    ln_proj_kernel<<<tilesL, 256, 0, stream>>>(
        l_embs1, NL, F(A2 + 12), F(A2 + 13),
        wq2p, F(A2 + 1), q_l,
        wk2p, F(A2 + 3), knode_l,
        wv2p, F(A2 + 5), vnode_l);

    edge_kv_kernel<7, true><<<tilesE2, 256, 0, stream>>>(
        l2l_ei, E2, l_position, l_heading, l_position, l_heading,
        hop, type3,
        F(L2LE + 0), F(L2LE + 1), F(L2LE + 2), F(L2LE + 3), w2_l2le, F(L2LE + 5),
        wke2p, F(A2 + 7), wve2p, F(A2 + 9),
        knode_l, vnode_l, kfull, vfull);

    zero_accums();
    {
        int nt = E2 * NH, gb = (nt + 255) / 256;
        score_kernel<<<gb, 256, 0, stream>>>(q_l, kfull, l2l_ei, E2, sc, mbuf);
        expsum_kernel<<<gb, 256, 0, stream>>>(sc, mbuf, l2l_ei, E2, den);
        agg_kernel<<<gb, 256, 0, stream>>>(sc, den, vfull, l2l_ei, E2, aggb);
    }
    attn_out_ffn_kernel<<<tilesL, 256, 0, stream>>>(
        l_embs1, aggb, NL,
        wo2p, F(A2 + 11), F(A2 + 14), F(A2 + 15),
        fw12p, F(A2 + 17), fw22p, F(A2 + 19),
        (float*)d_out);
}
